// BiGG_47124381171915
// MI455X (gfx1250) — matrix operand order repaired and checked by reading, not compiled, not run on hardware
//
#include <hip/hip_runtime.h>
#include <hip/hip_bf16.h>
#include <math.h>
#include <limits.h>

// ---------------------------------------------------------------------------
// BiGG forward on MI455X (gfx1250).
// GEMMs: v_wmma_f32_16x16x32_bf16 with both A and B pre-packed on device into
// per-lane-contiguous fragment layouts => all fragment traffic is
// global_load_b128.  Each wave computes a 16x64 strip (4 accumulators) so one
// A fragment feeds 4 WMMAs.  Edge softmax: fp32 VALU + global atomics.
// GRU: h + gate pre-activations in LDS, WMMA against packed W_ih^T / W_hh^T.
// ---------------------------------------------------------------------------

typedef unsigned int u32;
typedef __attribute__((ext_vector_type(16))) __bf16 v16bf;
typedef __attribute__((ext_vector_type(8)))  float  v8f;

union Frag {
    v16bf  v;
    u32    u[8];
    __bf16 e[16];
};
union Pair {
    __bf16 h[2];
    u32    u;
};

// ----- problem constants ----------------------------------------------------
#define NN    65536     // nodes
#define NE    65536     // edges
#define DEMB  128
#define HHEADS 4
#define FIN   32
#define BB    64
#define SS    64
#define LL    16
#define MROWS 4096      // B*S
#define RT_N  (NN / 16) // 4096 row tiles for the node GEMMs
#define SCALE_QK 0.08838834764831845f   // 1/sqrt(128)

// ===========================================================================
// small elementwise kernels
// ===========================================================================
__global__ void fill_f32_kernel(float* p, float v, int n) {
    int i = blockIdx.x * blockDim.x + threadIdx.x;
    if (i < n) p[i] = v;
}
__global__ void fill_i32_kernel(int* p, int v, int n) {
    int i = blockIdx.x * blockDim.x + threadIdx.x;
    if (i < n) p[i] = v;
}
__global__ void axpy_kernel(float* __restrict__ y, const float* __restrict__ x, float a, int n) {
    int i = blockIdx.x * blockDim.x + threadIdx.x;
    if (i < n) y[i] += a * x[i];
}

// ===========================================================================
// A-fragment pack: fp32 activations [M x ldx] -> bf16 WMMA A fragments.
// A-layout (ISA 7.12.2, 16-bit A 16x32): lane -> row m=L%16, half=L/16,
// element i -> K = (i<8 ? i : i+8) + 8*half; dword j holds K-pair
// koff(j) = (j<4) ? 2j+8*half : 2j+8+8*half.
// Output: dword j of lane L for (rowtile rt, kstep ks) at
//   [ ((rt*Ksteps + ks)*32 + L)*8 + j ]   (32 contiguous bytes per lane)
// ===========================================================================
__global__ void pack_a_kernel(const float* __restrict__ X, int ldx, int Ksteps,
                              int total, u32* __restrict__ out) {
    int t = blockIdx.x * blockDim.x + threadIdx.x;
    if (t >= total) return;
    int lane = t & 31;
    int ks   = (t >> 5) % Ksteps;
    int rt   = (t >> 5) / Ksteps;
    int half = lane >> 4;
    int row  = rt * 16 + (lane & 15);
    const float* xr = X + (size_t)row * ldx + ks * 32;
    u32* op = out + ((size_t)(rt * Ksteps + ks) * 32 + lane) * 8;
#pragma unroll
    for (int j = 0; j < 8; j++) {
        int koff = (j < 4) ? (2 * j + 8 * half) : (2 * j + 8 + 8 * half);
        Pair p;
        p.h[0] = (__bf16)xr[koff];
        p.h[1] = (__bf16)xr[koff + 1];
        op[j] = p.u;
    }
}

// GRU variant: A tile for (group g, step t) has rows (g*16 + ml)*LL + t of X.
// Output index: [ (((g*16 + t)*4 + ks)*32 + L)*8 + j ]
__global__ void pack_a_gru_kernel(const float* __restrict__ X, u32* __restrict__ out) {
    int t = blockIdx.x * blockDim.x + threadIdx.x;      // 256*16*4*32 threads
    if (t >= MROWS / 16 * LL * 4 * 32) return;
    int lane = t & 31;
    int ks   = (t >> 5) & 3;
    int tt   = (t >> 7) & 15;
    int g    = t >> 11;
    int half = lane >> 4;
    int row  = (g * 16 + (lane & 15)) * LL + tt;
    const float* xr = X + (size_t)row * DEMB + ks * 32;
    u32* op = out + ((size_t)((g * 16 + tt) * 4 + ks) * 32 + lane) * 8;
#pragma unroll
    for (int j = 0; j < 8; j++) {
        int koff = (j < 4) ? (2 * j + 8 * half) : (2 * j + 8 + 8 * half);
        Pair p;
        p.h[0] = (__bf16)xr[koff];
        p.h[1] = (__bf16)xr[koff + 1];
        op[j] = p.u;
    }
}

// ===========================================================================
// B-fragment pack: fp32 weights [K x N] (or transposed [N x K]) -> bf16.
// B-layout (ISA 7.12.2 / sparse-B pattern, 32x16): lane -> col n=L%16,
// half=L/16, element i -> K = (i<8 ? i : i+8) + 8*half.
// Output: dword j of lane L for (ks, nt) at [ ((ks*Ntiles+nt)*32 + L)*8 + j ]
// ===========================================================================
__global__ void pack_b_kernel(const float* __restrict__ W, int ldw, int col0,
                              int Ksteps, int Ntiles, int transposed,
                              u32* __restrict__ out) {
    int t = blockIdx.x * blockDim.x + threadIdx.x;
    int total = Ksteps * Ntiles * 32;
    if (t >= total) return;
    int lane = t & 31;
    int nt   = (t >> 5) % Ntiles;
    int ks   = (t >> 5) / Ntiles;
    int half = lane >> 4;
    int n    = col0 + nt * 16 + (lane & 15);
    Frag f;
#pragma unroll
    for (int i = 0; i < 16; i++) {
        int k = ks * 32 + ((i < 8) ? i : i + 8) + 8 * half;
        float v = transposed ? W[(size_t)n * ldw + k] : W[(size_t)k * ldw + n];
        f.e[i] = (__bf16)v;
    }
    u32* op = out + ((size_t)(ks * Ntiles + nt) * 32 + lane) * 8;
#pragma unroll
    for (int j = 0; j < 8; j++) op[j] = f.u[j];
}

// ===========================================================================
// WMMA GEMM, packed A and B, Ntiles == 8 (dout = 128).
// block = 64 threads = 2 waves; each wave owns a 16x64 strip (4 acc tiles);
// one A fragment per kstep feeds 4 WMMAs.  All fragment loads are b128.
// ===========================================================================
__global__ void wmma_gemm_kernel(const u32* __restrict__ Ap,
                                 const u32* __restrict__ Bp,
                                 const float* __restrict__ bias,
                                 float* __restrict__ C, int ldc,
                                 int Ksteps, int beta) {
    const int Ntiles = 8;
    int lane = threadIdx.x & 31;
    int wave = threadIdx.x >> 5;
    int rt   = blockIdx.x;
    int nt0  = wave * 4;
    int half = lane >> 4;

    v8f acc[4];
#pragma unroll
    for (int u = 0; u < 4; u++)
#pragma unroll
        for (int r = 0; r < 8; r++) acc[u][r] = 0.0f;

    for (int ks = 0; ks < Ksteps; ks++) {
        const uint4* ap = (const uint4*)(Ap + ((size_t)(rt * Ksteps + ks) * 32 + lane) * 8);
        Frag a;
        *(uint4*)&a.u[0] = ap[0];
        *(uint4*)&a.u[4] = ap[1];
        if (ks + 1 < Ksteps)   // gfx1250 global_prefetch_b8 for next k-step
            __builtin_prefetch(Ap + ((size_t)(rt * Ksteps + ks + 1) * 32 + lane) * 8, 0, 0);
#pragma unroll
        for (int u = 0; u < 4; u++) {
            const uint4* bp =
                (const uint4*)(Bp + ((size_t)(ks * Ntiles + nt0 + u) * 32 + lane) * 8);
            Frag b;
            *(uint4*)&b.u[0] = bp[0];
            *(uint4*)&b.u[4] = bp[1];
            acc[u] = __builtin_amdgcn_wmma_f32_16x16x32_bf16(
                false, a.v, false, b.v, (short)0, acc[u], false, false);
        }
    }

    int row0 = rt * 16;
#pragma unroll
    for (int u = 0; u < 4; u++) {
        int col = (nt0 + u) * 16 + (lane & 15);
        float bval = bias ? bias[col] : 0.0f;
        float* cp = C + col;
#pragma unroll
        for (int r = 0; r < 8; r++) {
            size_t row = (size_t)(row0 + r + 8 * half);
            float v = acc[u][r] + bval;
            if (beta) v += cp[row * ldc];
            cp[row * ldc] = v;
        }
    }
}

// ===========================================================================
// edge softmax phase (per head, per direction); ordered-int atomicMax trick.
// ===========================================================================
__device__ __forceinline__ int f2ord(float f) {
    int i = __float_as_int(f);
    return (i >= 0) ? i : (i ^ 0x7fffffff);
}
__device__ __forceinline__ float ord2f(int i) {
    return __int_as_float((i >= 0) ? i : (i ^ 0x7fffffff));
}

__global__ void edge_scores_kernel(const float* __restrict__ q, const float* __restrict__ k,
                                   const int* __restrict__ sidx, const int* __restrict__ didx,
                                   float* __restrict__ se, int* __restrict__ mx, int E) {
    int e = blockIdx.x * blockDim.x + threadIdx.x;
    if (e >= E) return;
    int s = sidx[e], d = didx[e];
    const float4* qp = (const float4*)(q + (size_t)d * DEMB);
    const float4* kp = (const float4*)(k + (size_t)s * DEMB);
    float acc = 0.0f;
#pragma unroll 8
    for (int i = 0; i < DEMB / 4; i++) {
        float4 a = qp[i], b = kp[i];
        acc += a.x * b.x + a.y * b.y + a.z * b.z + a.w * b.w;
    }
    float val = acc * SCALE_QK;
    se[e] = val;
    atomicMax(mx + d, f2ord(val));
}

__global__ void edge_expsum_kernel(float* __restrict__ se, const int* __restrict__ mx,
                                   float* __restrict__ z, const int* __restrict__ didx, int E) {
    int e = blockIdx.x * blockDim.x + threadIdx.x;
    if (e >= E) return;
    int d = didx[e];
    float m = ord2f(mx[d]);
    float ev = __expf(se[e] - m);
    se[e] = ev;
    atomicAdd(z + d, ev);
}

// one block (128 threads) per edge; scatter a*v into accum[dst]
__global__ void edge_scatter_kernel(const float* __restrict__ se, const float* __restrict__ z,
                                    const float* __restrict__ v,
                                    const int* __restrict__ sidx, const int* __restrict__ didx,
                                    float* __restrict__ accum) {
    int e = blockIdx.x;
    int c = threadIdx.x;
    int d = didx[e], s = sidx[e];
    float coef = se[e] / z[d];
    atomicAdd(accum + (size_t)d * DEMB + c, coef * v[(size_t)s * DEMB + c]);
}

// ===========================================================================
// per-feature stats over all N rows (reference normalizes over axis=0)
// ===========================================================================
__global__ void colstats_kernel(const float* __restrict__ x, float* __restrict__ mu,
                                float* __restrict__ rstd, int N) {
    __shared__ float s1[256];
    __shared__ float s2[256];
    int c = blockIdx.x;
    int tid = threadIdx.x;
    float a = 0.0f, b = 0.0f;
    for (int n = tid; n < N; n += 256) {
        float v = x[(size_t)n * DEMB + c];
        a += v;
        b += v * v;
    }
    s1[tid] = a; s2[tid] = b;
    __syncthreads();
    for (int st = 128; st > 0; st >>= 1) {
        if (tid < st) { s1[tid] += s1[tid + st]; s2[tid] += s2[tid + st]; }
        __syncthreads();
    }
    if (tid == 0) {
        float m = s1[0] / (float)N;
        float var = s2[0] / (float)N - m * m;
        mu[c] = m;
        rstd[c] = rsqrtf(var + 1e-5f);
    }
}

__global__ void ln_relu_kernel(const float* __restrict__ xn, const float* __restrict__ mu,
                               const float* __restrict__ rstd, const float* __restrict__ gamma,
                               const float* __restrict__ beta_, float* __restrict__ xout, int n) {
    int i = blockIdx.x * blockDim.x + threadIdx.x;
    if (i >= n) return;
    int c = i & (DEMB - 1);
    float v = (xn[i] - mu[c]) * rstd[c] * gamma[c] + beta_[c];
    xout[i] = fmaxf(v, 0.0f);
}

// ===========================================================================
// GRU: one wave per block, 16 sequences per block.  h and gate pre-activations
// in LDS; x_t fragments pre-packed (b128 loads); W_ih^T/W_hh^T pre-packed.
// ===========================================================================
__global__ void __launch_bounds__(32)
gru_kernel(const u32* __restrict__ Xp, const u32* __restrict__ Wih_p,
           const u32* __restrict__ Whh_p, const float* __restrict__ b_ih,
           const float* __restrict__ b_hh, float* __restrict__ hout) {
    __shared__ float h_lds[16 * 128];    // 8 KB
    __shared__ float gi_lds[16 * 384];   // 24 KB
    __shared__ float gh_lds[16 * 384];   // 24 KB
    int lane = threadIdx.x;
    int g    = blockIdx.x;
    int m0   = g * 16;
    int ml   = lane & 15;
    int half = lane >> 4;

    for (int i = lane; i < 16 * 128; i += 32) h_lds[i] = 0.0f;
    __syncthreads();

    for (int t = 0; t < LL; t++) {
        // x_t A-fragments: packed, 2 x b128 per kstep
        Frag ax[4];
#pragma unroll
        for (int ks = 0; ks < 4; ks++) {
            const uint4* ap =
                (const uint4*)(Xp + ((size_t)((g * 16 + t) * 4 + ks) * 32 + lane) * 8);
            *(uint4*)&ax[ks].u[0] = ap[0];
            *(uint4*)&ax[ks].u[4] = ap[1];
        }
        // h A-fragments from LDS (fp32 -> bf16)
        Frag ah[4];
#pragma unroll
        for (int ks = 0; ks < 4; ks++) {
#pragma unroll
            for (int i = 0; i < 16; i++) {
                int k = ks * 32 + ((i < 8) ? i : i + 8) + 8 * half;
                ah[ks].e[i] = (__bf16)h_lds[ml * 128 + k];
            }
        }
        // gi = x_t @ W_ih^T + b_ih ; gh = h @ W_hh^T + b_hh   (24 col tiles)
        for (int nt = 0; nt < 24; nt++) {
            v8f ci, ch;
#pragma unroll
            for (int r = 0; r < 8; r++) { ci[r] = 0.0f; ch[r] = 0.0f; }
#pragma unroll
            for (int ks = 0; ks < 4; ks++) {
                Frag b;
                const uint4* bp =
                    (const uint4*)(Wih_p + ((size_t)(ks * 24 + nt) * 32 + lane) * 8);
                *(uint4*)&b.u[0] = bp[0];
                *(uint4*)&b.u[4] = bp[1];
                ci = __builtin_amdgcn_wmma_f32_16x16x32_bf16(
                    false, ax[ks].v, false, b.v, (short)0, ci, false, false);
                const uint4* bp2 =
                    (const uint4*)(Whh_p + ((size_t)(ks * 24 + nt) * 32 + lane) * 8);
                *(uint4*)&b.u[0] = bp2[0];
                *(uint4*)&b.u[4] = bp2[1];
                ch = __builtin_amdgcn_wmma_f32_16x16x32_bf16(
                    false, ah[ks].v, false, b.v, (short)0, ch, false, false);
            }
            int col = nt * 16 + ml;
            float bi = b_ih[col], bh = b_hh[col];
#pragma unroll
            for (int r = 0; r < 8; r++) {
                int rowl = r + 8 * half;
                gi_lds[rowl * 384 + col] = ci[r] + bi;
                gh_lds[rowl * 384 + col] = ch[r] + bh;
            }
        }
        __syncthreads();
        // gate update
        for (int idx = lane; idx < 16 * 128; idx += 32) {
            int mm = idx >> 7, d = idx & 127;
            float ir = gi_lds[mm * 384 + d];
            float iz = gi_lds[mm * 384 + 128 + d];
            float in_ = gi_lds[mm * 384 + 256 + d];
            float hr = gh_lds[mm * 384 + d];
            float hz = gh_lds[mm * 384 + 128 + d];
            float hn = gh_lds[mm * 384 + 256 + d];
            float r  = 1.0f / (1.0f + __expf(-(ir + hr)));
            float zt = 1.0f / (1.0f + __expf(-(iz + hz)));
            float nt_ = tanhf(in_ + r * hn);
            h_lds[idx] = (1.0f - zt) * nt_ + zt * h_lds[idx];
        }
        __syncthreads();
    }
    for (int idx = lane; idx < 16 * 128; idx += 32) {
        int mm = idx >> 7, d = idx & 127;
        hout[(size_t)(m0 + mm) * DEMB + d] = h_lds[idx];
    }
}

// ===========================================================================
// output assembly: y[64,128], M[4096,256], gmax[64], local[4096] concat flat
// ===========================================================================
__global__ void outputs_kernel(const float* __restrict__ h, const float* __restrict__ labels,
                               float* __restrict__ out) {
    const int NY = BB * DEMB;            // 8192
    const int NM = MROWS * 2 * DEMB;     // 1048576
    const int NG = BB;                   // 64
    const int NL = MROWS;                // 4096
    int i = blockIdx.x * blockDim.x + threadIdx.x;
    if (i < NY) {
        int b = i >> 7, d = i & 127;
        out[i] = h[(size_t)(b * SS + SS - 1) * DEMB + d];
    } else if (i < NY + NM) {
        int j = i - NY;
        int row = j >> 8, c = j & 255;
        int b = row >> 6;
        float v = (c < DEMB) ? h[(size_t)row * DEMB + c]
                             : h[(size_t)(b * SS + SS - 1) * DEMB + (c - DEMB)];
        out[i] = v;
    } else if (i < NY + NM + NG) {
        int b = i - NY - NM;
        out[i] = labels[b * SS + SS - 1];
    } else if (i < NY + NM + NG + NL) {
        int j = i - NY - NM - NG;
        int b = j >> 6;
        out[i] = labels[j] / labels[b * SS + SS - 1];
    }
}

// ===========================================================================
// host orchestration
// ===========================================================================
static inline int cdiv(int a, int b) { return (a + b - 1) / b; }

extern "C" void kernel_launch(void* const* d_in, const int* in_sizes, int n_in,
                              void* d_out, int out_size, void* d_ws, size_t ws_size,
                              hipStream_t stream) {
    (void)in_sizes; (void)n_in; (void)out_size; (void)ws_size;

    // ---- input map -------------------------------------------------------
    // top-level in setup_inputs() insertion order; params flattened as a jax
    // pytree (dict keys sorted): gru{W_hh,W_ih,b_hh,b_ih}, then layers[0..1]
    // each {Wr,beta,br,gamma,in{Wk,Wq,Wv,bk,bq,bv},out{Wk,Wq,Wv,bk,bq,bv}}
    const float* node_feat = (const float*)d_in[0];
    const int*   e0 = (const int*)d_in[1];          // src
    const int*   e1 = e0 + NE;                      // dst
    const float* labels = (const float*)d_in[2];
    const float* gru_Whh = (const float*)d_in[3];
    const float* gru_Wih = (const float*)d_in[4];
    const float* gru_bhh = (const float*)d_in[5];
    const float* gru_bih = (const float*)d_in[6];

    // ---- workspace -------------------------------------------------------
    char* wsp = (char*)d_ws;
    size_t off = 0;
    auto walloc = [&](size_t bytes) -> void* {
        void* p = wsp + off;
        off = (off + bytes + 255) & ~(size_t)255;
        return p;
    };
    u32*    Ap    = (u32*)walloc((size_t)RT_N * 4 * 32 * 8 * 4);   // 16 MB (also GRU Xp)
    float*  xf    = (float*)walloc((size_t)NN * DEMB * 4);
    float*  xnext = (float*)walloc((size_t)NN * DEMB * 4);
    float*  qbuf  = (float*)walloc((size_t)NN * DEMB * 4);         // reused as accum
    float*  kbuf  = (float*)walloc((size_t)NN * DEMB * 4);
    float*  vbuf  = (float*)walloc((size_t)NN * DEMB * 4);
    float*  se    = (float*)walloc((size_t)NE * 4);
    int*    mx    = (int*)walloc((size_t)NN * 4);
    float*  zb    = (float*)walloc((size_t)NN * 4);
    float*  mu    = (float*)walloc(DEMB * 4);
    float*  rstd  = (float*)walloc(DEMB * 4);
    u32*    Wp0   = (u32*)walloc(128 * 1024);
    u32*    Wp1   = (u32*)walloc(128 * 1024);
    u32*    Wp2   = (u32*)walloc(128 * 1024);
    float*  hout  = (float*)walloc((size_t)MROWS * DEMB * 4);

    const int TPB = 256;
    const int ND  = NN * DEMB;   // 8,388,608

    const float* xcur = node_feat;
    int din = FIN;

    for (int li = 0; li < 2; li++) {
        int base = 7 + li * 16;
        const float* Wr     = (const float*)d_in[base + 0];
        const float* beta_p = (const float*)d_in[base + 1];
        const float* br     = (const float*)d_in[base + 2];
        const float* gamma  = (const float*)d_in[base + 3];
        int Ksteps = din / 32;

        // pack all A fragments for this layer (fp32 -> bf16 fused)
        int aTot = RT_N * Ksteps * 32;
        pack_a_kernel<<<cdiv(aTot, TPB), TPB, 0, stream>>>(xcur, din, Ksteps, aTot, Ap);
        // x_next = 0
        fill_f32_kernel<<<cdiv(ND, TPB), TPB, 0, stream>>>(xnext, 0.0f, ND);

        // x_next += x @ Wr + br   (Wr: [din,128] -> 8 col tiles)
        pack_b_kernel<<<cdiv(Ksteps * 8 * 32, 128), 128, 0, stream>>>(
            Wr, DEMB, 0, Ksteps, 8, 0, Wp0);
        wmma_gemm_kernel<<<RT_N, 64, 0, stream>>>(Ap, Wp0, br, xnext, DEMB, Ksteps, 1);

        for (int dir = 0; dir < 2; dir++) {
            // dir 0 = 'in'  conv: messages src->dst  (src=e0, dst=e1)
            // dir 1 = 'out' conv: messages dst->src  (src=e1, dst=e0)
            const int* sidx = dir ? e1 : e0;
            const int* didx = dir ? e0 : e1;
            int pb = base + 4 + dir * 6;     // Wk,Wq,Wv,bk,bq,bv
            const float* Wk = (const float*)d_in[pb + 0];
            const float* Wq = (const float*)d_in[pb + 1];
            const float* Wv = (const float*)d_in[pb + 2];
            const float* bk = (const float*)d_in[pb + 3];
            const float* bq = (const float*)d_in[pb + 4];
            const float* bv = (const float*)d_in[pb + 5];

            for (int h = 0; h < HHEADS; h++) {
                int col0 = h * DEMB;
                int packThreads = Ksteps * 8 * 32;
                pack_b_kernel<<<cdiv(packThreads, 128), 128, 0, stream>>>(
                    Wq, HHEADS * DEMB, col0, Ksteps, 8, 0, Wp0);
                pack_b_kernel<<<cdiv(packThreads, 128), 128, 0, stream>>>(
                    Wk, HHEADS * DEMB, col0, Ksteps, 8, 0, Wp1);
                pack_b_kernel<<<cdiv(packThreads, 128), 128, 0, stream>>>(
                    Wv, HHEADS * DEMB, col0, Ksteps, 8, 0, Wp2);
                wmma_gemm_kernel<<<RT_N, 64, 0, stream>>>(
                    Ap, Wp0, bq + col0, qbuf, DEMB, Ksteps, 0);
                wmma_gemm_kernel<<<RT_N, 64, 0, stream>>>(
                    Ap, Wp1, bk + col0, kbuf, DEMB, Ksteps, 0);
                wmma_gemm_kernel<<<RT_N, 64, 0, stream>>>(
                    Ap, Wp2, bv + col0, vbuf, DEMB, Ksteps, 0);

                // segment softmax over incoming edges of each dst
                fill_i32_kernel<<<cdiv(NN, TPB), TPB, 0, stream>>>(mx, INT_MIN, NN);
                fill_f32_kernel<<<cdiv(NN, TPB), TPB, 0, stream>>>(zb, 0.0f, NN);
                edge_scores_kernel<<<cdiv(NE, TPB), TPB, 0, stream>>>(
                    qbuf, kbuf, sidx, didx, se, mx, NE);
                edge_expsum_kernel<<<cdiv(NE, TPB), TPB, 0, stream>>>(se, mx, zb, didx, NE);

                // scatter a*v into accum (reuse qbuf; q no longer needed)
                fill_f32_kernel<<<cdiv(ND, TPB), TPB, 0, stream>>>(qbuf, 0.0f, ND);
                edge_scatter_kernel<<<NE, DEMB, 0, stream>>>(se, zb, vbuf, sidx, didx, qbuf);

                // x_next += w_dir * (1/H) * accum ;  w_dir = 0.5 both dirs
                axpy_kernel<<<cdiv(ND, TPB), TPB, 0, stream>>>(xnext, qbuf, 0.125f, ND);
            }
        }

        // per-feature normalization over nodes, then ReLU -> xf
        colstats_kernel<<<DEMB, 256, 0, stream>>>(xnext, mu, rstd, NN);
        ln_relu_kernel<<<cdiv(ND, TPB), TPB, 0, stream>>>(
            xnext, mu, rstd, gamma, beta_p, xf, ND);

        xcur = xf;
        din = DEMB;
    }

    // ---- GRU over [4096, 16, 128] ---------------------------------------
    int gTot = (MROWS / 16) * LL * 4 * 32;
    pack_a_gru_kernel<<<cdiv(gTot, TPB), TPB, 0, stream>>>(xf, Ap);
    // B = W_ih^T / W_hh^T : [128 x 384]  (transposed read of [384,128])
    pack_b_kernel<<<cdiv(4 * 24 * 32, 128), 128, 0, stream>>>(
        gru_Wih, DEMB, 0, 4, 24, 1, Wp0);
    pack_b_kernel<<<cdiv(4 * 24 * 32, 128), 128, 0, stream>>>(
        gru_Whh, DEMB, 0, 4, 24, 1, Wp1);
    gru_kernel<<<MROWS / 16, 32, 0, stream>>>(Ap, Wp0, Wp1, gru_bih, gru_bhh, hout);

    // ---- outputs ---------------------------------------------------------
    const int OUT_TOTAL = BB * DEMB + MROWS * 2 * DEMB + BB + MROWS;  // 1060928
    outputs_kernel<<<cdiv(OUT_TOTAL, TPB), TPB, 0, stream>>>(
        hout, labels, (float*)d_out);
}
